// PatternsOfThinking_30623116820925
// MI455X (gfx1250) — matrix operand order repaired and checked by reading, not compiled, not run on hardware
//
#include <hip/hip_runtime.h>
#include <hip/hip_bf16.h>
#include <math.h>

// ---------------------------------------------------------------------------
// CDNA5 / gfx1250 fused transformer block (B=2,S=2048,E=1024,H=16,D=64)
// Heavy matmuls: v_wmma_f32_16x16x32_bf16 (wave32).
// Tile staging:  global_load_async_to_lds_b128 (+ s_wait_asynccnt), double-buffered.
// ---------------------------------------------------------------------------

typedef __bf16 bf16_t;
typedef __attribute__((ext_vector_type(16))) __bf16 v16bf;
typedef __attribute__((ext_vector_type(8)))  __bf16 v8bf;
typedef __attribute__((ext_vector_type(8)))  float  v8f;

#define B_ 2
#define S_ 2048
#define E_ 1024
#define H_ 16
#define D_ 64
#define EPS_ 1e-5f

// async copy global->LDS, 16 bytes per lane, ASYNCcnt-tracked
#define ASYNC_B128(ldsaddr, goff, base)                                   \
    asm volatile("global_load_async_to_lds_b128 %0, %1, %2"               \
                 :: "v"(ldsaddr), "v"(goff), "s"(base) : "memory")
#define WAIT_ASYNCCNT(n) asm volatile("s_wait_asynccnt " #n ::: "memory")

__device__ __forceinline__ float gelu_exact(float x) {
    return 0.5f * x * (1.0f + erff(x * 0.70710678118654752f));
}

__device__ __forceinline__ v16bf cat8(v8bf lo, v8bf hi) {
    return __builtin_shufflevector(lo, hi, 0,1,2,3,4,5,6,7,8,9,10,11,12,13,14,15);
}

__device__ __forceinline__ unsigned lds_addr(const void* p) {
    return (unsigned)(size_t)p;   // low 32 bits of generic ptr == LDS offset
}

// ---------------------------------------------------------------------------
// f32 -> bf16 conversion
// ---------------------------------------------------------------------------
__global__ __launch_bounds__(256) void cvt_f32_bf16(const float* __restrict__ in,
                                                    bf16_t* __restrict__ out, int n) {
    int idx = (blockIdx.x * 256 + threadIdx.x) * 4;
    if (idx < n) {
        float4 v = *(const float4*)(in + idx);
        out[idx+0] = (bf16_t)v.x;
        out[idx+1] = (bf16_t)v.y;
        out[idx+2] = (bf16_t)v.z;
        out[idx+3] = (bf16_t)v.w;
    }
}

// ---------------------------------------------------------------------------
// GEMM: out[M,N] = act( A[M,K] @ W[N,K]^T + bias[N] )
// block 256 thr (8 waves), block tile 128x128, wave tile 32x64.
// Double-buffered async LDS staging, 32-deep k-stages.
// LDS rows padded to 40 bf16 (80B = 20 DW stride -> 16 distinct banks).
// ---------------------------------------------------------------------------
#define LDP 40

template<bool GELU, bool OUT_BF16, bool OUT_F32>
__global__ __launch_bounds__(256)
void gemm_bf16(const bf16_t* __restrict__ A, const bf16_t* __restrict__ W,
               const float* __restrict__ bias,
               bf16_t* __restrict__ Obf, float* __restrict__ Of,
               int M, int N, int K) {
    __shared__ __align__(16) bf16_t Abuf[2][128][LDP];
    __shared__ __align__(16) bf16_t Bbuf[2][128][LDP];
    const int tid  = threadIdx.x;
    const int lane = tid & 31, wave = tid >> 5;
    const int wr = wave & 3, wc = wave >> 2;          // 4x2 wave grid
    const int row0 = blockIdx.y * 128, col0 = blockIdx.x * 128;
    const int hn = lane & 15, half = lane >> 4;

    // staging: 128x32 bf16 tile = 512 x 16B chunks = 2 chunks/thread/array
    const int srow = tid >> 2;            // 0..63 base row (x2 chunks -> 128 rows)
    const int sc8  = (tid & 3) << 3;      // 16B chunk column (0,8,16,24)
    unsigned aoff[2], boff[2], lA[2][2], lB[2][2];
    #pragma unroll
    for (int j = 0; j < 2; ++j) {
        const int r = srow + 64 * j;
        aoff[j] = (unsigned)(((size_t)(row0 + r) * K + sc8) * 2);
        boff[j] = (unsigned)(((size_t)(col0 + r) * K + sc8) * 2);
        #pragma unroll
        for (int bi = 0; bi < 2; ++bi) {
            lA[bi][j] = lds_addr(&Abuf[bi][r][sc8]);
            lB[bi][j] = lds_addr(&Bbuf[bi][r][sc8]);
        }
    }
    const unsigned long long Ab64 = (unsigned long long)(size_t)A;
    const unsigned long long Wb64 = (unsigned long long)(size_t)W;

    // prologue: stage 0 into buffer 0
    #pragma unroll
    for (int j = 0; j < 2; ++j) ASYNC_B128(lA[0][j], aoff[j], Ab64);
    #pragma unroll
    for (int j = 0; j < 2; ++j) ASYNC_B128(lB[0][j], boff[j], Wb64);

    v8f acc[2][4] = {};
    const int nk = K / 32;
    for (int kk = 0; kk < nk; ++kk) {
        if (kk + 1 < nk) {
            const int nb = (kk + 1) & 1;
            const unsigned kb = (unsigned)((kk + 1) * 64);   // 32 bf16 = 64 bytes
            #pragma unroll
            for (int j = 0; j < 2; ++j) ASYNC_B128(lA[nb][j], aoff[j] + kb, Ab64);
            #pragma unroll
            for (int j = 0; j < 2; ++j) ASYNC_B128(lB[nb][j], boff[j] + kb, Wb64);
            WAIT_ASYNCCNT(4);     // previous stage (4 loads) complete
        } else {
            WAIT_ASYNCCNT(0);
        }
        __syncthreads();
        const int cb = kk & 1;
        v16bf af[2];
        #pragma unroll
        for (int rt = 0; rt < 2; ++rt) {
            const bf16_t* ar = &Abuf[cb][wr * 32 + rt * 16 + hn][0];
            af[rt] = cat8(*(const v8bf*)(ar + 8 * half),
                          *(const v8bf*)(ar + 16 + 8 * half));
        }
        #pragma unroll
        for (int ct = 0; ct < 4; ++ct) {
            const bf16_t* br = &Bbuf[cb][wc * 64 + ct * 16 + hn][0];
            v16bf bfr = cat8(*(const v8bf*)(br + 8 * half),
                             *(const v8bf*)(br + 16 + 8 * half));
            #pragma unroll
            for (int rt = 0; rt < 2; ++rt)
                acc[rt][ct] = __builtin_amdgcn_wmma_f32_16x16x32_bf16(
                    false, af[rt], false, bfr, (short)0, acc[rt][ct], false, false);
        }
        __syncthreads();
    }

    // epilogue: C layout -> row = r + 8*half, col = lane&15
    #pragma unroll
    for (int rt = 0; rt < 2; ++rt)
    #pragma unroll
    for (int ct = 0; ct < 4; ++ct) {
        const int coln = col0 + wc * 64 + ct * 16 + hn;
        const float bv = bias[coln];
        #pragma unroll
        for (int r = 0; r < 8; ++r) {
            const int rowm = row0 + wr * 32 + rt * 16 + r + 8 * half;
            float v = acc[rt][ct][r] + bv;
            if (GELU) v = gelu_exact(v);
            const size_t o = (size_t)rowm * N + coln;
            if (OUT_BF16) Obf[o] = (bf16_t)v;
            if (OUT_F32)  Of[o]  = v;
        }
    }
}

// ---------------------------------------------------------------------------
// Flash attention with argmax tracking.
// grid = (S/128, B*H); block = 256 (8 waves x 16 query rows).
// K tile staged via async-to-LDS; V staged transposed; padded strides.
// ---------------------------------------------------------------------------
__global__ __launch_bounds__(256)
void flash_attn(const bf16_t* __restrict__ Qm, const bf16_t* __restrict__ Km,
                const bf16_t* __restrict__ Vm,
                float* __restrict__ O, float* __restrict__ vals,
                int* __restrict__ amax) {
    __shared__ __align__(16) bf16_t Kt[32][88];     // 32 keys x 64 d (pad->88)
    __shared__ __align__(16) bf16_t VT[64][40];     // V^T: [d][t]   (pad->40)
    __shared__ __align__(16) bf16_t Pl[8][16][40];  // per-wave P    (pad->40)

    const int tid  = threadIdx.x;
    const int lane = tid & 31, wave = tid >> 5;
    const int hn = lane & 15, half = lane >> 4;
    const int bh = blockIdx.y;
    const int b = bh >> 4, h = bh & 15;
    const int q0 = blockIdx.x * 128 + wave * 16;

    // Q fragments resident in registers: D=64 -> 2 k-steps of 32
    const bf16_t* qbase = Qm + (size_t)(b * S_ + q0 + hn) * E_ + h * D_;
    v16bf qa0 = cat8(*(const v8bf*)(qbase + 8 * half),
                     *(const v8bf*)(qbase + 16 + 8 * half));
    v16bf qa1 = cat8(*(const v8bf*)(qbase + 32 + 8 * half),
                     *(const v8bf*)(qbase + 48 + 8 * half));

    // async K staging: thread -> row tid>>3, 16B chunk (tid&7)*8
    const int srow = tid >> 3, sc8 = (tid & 7) << 3;
    const unsigned ldsK = lds_addr(&Kt[srow][sc8]);
    unsigned goffK = (unsigned)(((size_t)(b * S_ + srow) * E_ + h * D_ + sc8) * 2);
    const unsigned long long Kb64 = (unsigned long long)(size_t)Km;

    float m_[8], l_[8];
    int   ix[8];
    #pragma unroll
    for (int r = 0; r < 8; ++r) { m_[r] = -INFINITY; l_[r] = 0.f; ix[r] = 0; }
    v8f o[4] = {};
    const float inv_sqrt_e = 0.03125f;   // 1/sqrt(1024)

    for (int t0 = 0; t0 < S_; t0 += 32) {
        ASYNC_B128(ldsK, goffK, Kb64);
        goffK += 32 * E_ * 2;
        // V tile transposed (manual: element transpose during store)
        {
            int t = tid >> 3, d0 = (tid & 7) << 3;
            v8bf v = *(const v8bf*)&Vm[(size_t)(b * S_ + t0 + t) * E_ + h * D_ + d0];
            #pragma unroll
            for (int j = 0; j < 8; ++j) VT[d0 + j][t] = v[j];
        }
        WAIT_ASYNCCNT(0);
        __syncthreads();

        // scores S[16x32] = Q(16x64) @ K^T, two 16-col tiles
        v8f s[2];
        #pragma unroll
        for (int nt = 0; nt < 2; ++nt) {
            const int kn = nt * 16 + hn;
            v16bf bk0 = cat8(*(const v8bf*)&Kt[kn][8 * half],
                             *(const v8bf*)&Kt[kn][16 + 8 * half]);
            v16bf bk1 = cat8(*(const v8bf*)&Kt[kn][32 + 8 * half],
                             *(const v8bf*)&Kt[kn][48 + 8 * half]);
            v8f c = {};
            c = __builtin_amdgcn_wmma_f32_16x16x32_bf16(false, qa0, false, bk0,
                                                        (short)0, c, false, false);
            c = __builtin_amdgcn_wmma_f32_16x16x32_bf16(false, qa1, false, bk1,
                                                        (short)0, c, false, false);
            s[nt] = c;
        }

        float scale[8];
        #pragma unroll
        for (int r = 0; r < 8; ++r) {
            float v0 = s[0][r] * inv_sqrt_e;
            float v1 = s[1][r] * inv_sqrt_e;
            float tv = v0; int ti = t0 + hn;
            if (v1 > tv) { tv = v1; ti = t0 + 16 + hn; }
            #pragma unroll
            for (int msk = 1; msk < 16; msk <<= 1) {
                float ov = __shfl_xor(tv, msk, 32);
                int   oi = __shfl_xor(ti, msk, 32);
                if (ov > tv || (ov == tv && oi < ti)) { tv = ov; ti = oi; }
            }
            float newm = m_[r];
            if (tv > newm) { newm = tv; ix[r] = ti; }
            scale[r] = __expf(m_[r] - newm);
            m_[r] = newm;
            float p0 = __expf(v0 - newm);
            float p1 = __expf(v1 - newm);
            float ps = p0 + p1;
            #pragma unroll
            for (int msk = 1; msk < 16; msk <<= 1) ps += __shfl_xor(ps, msk, 32);
            l_[r] = l_[r] * scale[r] + ps;
            Pl[wave][r + 8 * half][hn]      = (bf16_t)p0;
            Pl[wave][r + 8 * half][16 + hn] = (bf16_t)p1;
        }

        #pragma unroll
        for (int dt = 0; dt < 4; ++dt)
            #pragma unroll
            for (int r = 0; r < 8; ++r)
                o[dt][r] = o[dt][r] * scale[r];

        // O += P @ V  (P as A-frag, V^T rows as B-frags)
        v16bf pa = cat8(*(const v8bf*)&Pl[wave][hn][8 * half],
                        *(const v8bf*)&Pl[wave][hn][16 + 8 * half]);
        #pragma unroll
        for (int dt = 0; dt < 4; ++dt) {
            const int dn = dt * 16 + hn;
            v16bf bv = cat8(*(const v8bf*)&VT[dn][8 * half],
                            *(const v8bf*)&VT[dn][16 + 8 * half]);
            o[dt] = __builtin_amdgcn_wmma_f32_16x16x32_bf16(false, pa, false, bv,
                                                            (short)0, o[dt], false, false);
        }
        __syncthreads();
    }

    #pragma unroll
    for (int dt = 0; dt < 4; ++dt)
        #pragma unroll
        for (int r = 0; r < 8; ++r) {
            const int row = q0 + r + 8 * half;
            O[(size_t)(b * S_ + row) * E_ + h * D_ + dt * 16 + hn] = o[dt][r] / l_[r];
        }
    if (hn == 0) {
        #pragma unroll
        for (int r = 0; r < 8; ++r) {
            const int row = q0 + r + 8 * half;
            vals[(size_t)bh * S_ + row] = 1.0f / l_[r];
            amax[(size_t)bh * S_ + row] = ix[r];
        }
    }
}

// ---------------------------------------------------------------------------
// Pot-block Linear over seq dim: arr = gelu(vals @ Wp^T + bp)   [32x2048 rows]
// ---------------------------------------------------------------------------
__global__ __launch_bounds__(256)
void pot_gemm(const float* __restrict__ vals, const float* __restrict__ Wp,
              const float* __restrict__ bp, float* __restrict__ arr) {
    __shared__ float vrow[S_];
    const int bh = blockIdx.x;
    const float* v = vals + (size_t)bh * S_;
    for (int i = threadIdx.x; i < S_; i += 256) vrow[i] = v[i];
    __syncthreads();
    for (int jj = 0; jj < 8; ++jj) {
        const int j = jj * 256 + threadIdx.x;
        const float* wr_ = Wp + (size_t)j * S_;
        float accv = 0.f;
        for (int k = 0; k < S_; k += 4) {
            float4 w4 = *(const float4*)(wr_ + k);
            accv += vrow[k] * w4.x + vrow[k+1] * w4.y +
                    vrow[k+2] * w4.z + vrow[k+3] * w4.w;
        }
        arr[(size_t)bh * S_ + j] = gelu_exact(accv + bp[j]);
    }
}

// ---------------------------------------------------------------------------
// Rank-1 argmax-overwrite correction; also emits bf16 O for the FFN.
// ---------------------------------------------------------------------------
__global__ __launch_bounds__(256)
void pot_correct(const float* __restrict__ vals, const float* __restrict__ arr,
                 const int* __restrict__ amax, const bf16_t* __restrict__ Vm,
                 const float* __restrict__ O, bf16_t* __restrict__ Ob) {
    const size_t gid = (size_t)blockIdx.x * 256 + threadIdx.x;
    const int d = gid & 63;
    const size_t row = gid >> 6;           // [b][h][s]
    const int s = (int)(row & (S_ - 1));
    const int h = (int)((row >> 11) & 15);
    const int b = (int)(row >> 15);
    const float delta = arr[row] - vals[row];
    const int t = amax[row];
    const float vv = (float)Vm[(size_t)(b * S_ + t) * E_ + h * D_ + d];
    const size_t oidx = (size_t)(b * S_ + s) * E_ + h * D_ + d;
    Ob[oidx] = (bf16_t)(O[oidx] + delta * vv);
}

// ---------------------------------------------------------------------------
// LayerNorm over E=1024; one block (256 thr) per row.
// ---------------------------------------------------------------------------
__global__ __launch_bounds__(256)
void layernorm(const float* __restrict__ hsrc, const float* __restrict__ gamma,
               const float* __restrict__ beta, float* __restrict__ out) {
    __shared__ float s1[8], s2[8];
    const int row = blockIdx.x;
    const float* hr = hsrc + (size_t)row * E_;
    float vloc[4], sum = 0.f, sq = 0.f;
    #pragma unroll
    for (int j = 0; j < 4; ++j) {
        float v = hr[threadIdx.x + j * 256];
        vloc[j] = v; sum += v; sq += v * v;
    }
    #pragma unroll
    for (int msk = 1; msk < 32; msk <<= 1) {
        sum += __shfl_xor(sum, msk, 32);
        sq  += __shfl_xor(sq,  msk, 32);
    }
    const int wave = threadIdx.x >> 5, lane = threadIdx.x & 31;
    if (lane == 0) { s1[wave] = sum; s2[wave] = sq; }
    __syncthreads();
    if (wave == 0) {
        float a = (lane < 8) ? s1[lane] : 0.f;
        float c = (lane < 8) ? s2[lane] : 0.f;
        #pragma unroll
        for (int msk = 1; msk < 8; msk <<= 1) {
            a += __shfl_xor(a, msk, 32);
            c += __shfl_xor(c, msk, 32);
        }
        if (lane == 0) { s1[0] = a; s2[0] = c; }
    }
    __syncthreads();
    const float mu   = s1[0] * (1.f / E_);
    const float var  = s2[0] * (1.f / E_) - mu * mu;
    const float rinv = rsqrtf(var + EPS_);
    #pragma unroll
    for (int j = 0; j < 4; ++j) {
        const int c = threadIdx.x + j * 256;
        out[(size_t)row * E_ + c] = gamma[c] * (vloc[j] - mu) * rinv + beta[c];
    }
}

// ---------------------------------------------------------------------------
extern "C" void kernel_launch(void* const* d_in, const int* in_sizes, int n_in,
                              void* d_out, int out_size, void* d_ws, size_t ws_size,
                              hipStream_t stream) {
    (void)in_sizes; (void)n_in; (void)out_size; (void)ws_size;
    const float* x     = (const float*)d_in[0];
    const float* Wq    = (const float*)d_in[1];
    const float* bq    = (const float*)d_in[2];
    const float* Wk    = (const float*)d_in[3];
    const float* bk    = (const float*)d_in[4];
    const float* Wv    = (const float*)d_in[5];
    const float* bv    = (const float*)d_in[6];
    const float* Wp    = (const float*)d_in[7];
    const float* bp    = (const float*)d_in[8];
    const float* W1    = (const float*)d_in[9];
    const float* b1    = (const float*)d_in[10];
    const float* W2    = (const float*)d_in[11];
    const float* b2    = (const float*)d_in[12];
    const float* gamma = (const float*)d_in[13];
    const float* beta  = (const float*)d_in[14];
    float* outp = (float*)d_out;

    char* ws = (char*)d_ws;
    size_t off = 0;
    auto nxt = [&](size_t bytes) -> char* {
        char* p = ws + off;
        off += (bytes + 255) & ~(size_t)255;
        return p;
    };
    const int MS = B_ * S_;  // 4096 rows
    bf16_t* xb   = (bf16_t*)nxt((size_t)MS * E_ * 2);
    bf16_t* Wqb  = (bf16_t*)nxt((size_t)E_ * E_ * 2);
    bf16_t* Wkb  = (bf16_t*)nxt((size_t)E_ * E_ * 2);
    bf16_t* Wvb  = (bf16_t*)nxt((size_t)E_ * E_ * 2);
    bf16_t* W1b  = (bf16_t*)nxt((size_t)4 * E_ * E_ * 2);
    bf16_t* W2b  = (bf16_t*)nxt((size_t)4 * E_ * E_ * 2);
    bf16_t* Qb   = (bf16_t*)nxt((size_t)MS * E_ * 2);
    bf16_t* Kb   = (bf16_t*)nxt((size_t)MS * E_ * 2);
    bf16_t* Vb   = (bf16_t*)nxt((size_t)MS * E_ * 2);
    float*  Of   = (float*) nxt((size_t)MS * E_ * 4);
    bf16_t* Obf  = (bf16_t*)nxt((size_t)MS * E_ * 2);
    float*  valsb= (float*) nxt((size_t)B_ * H_ * S_ * 4);
    float*  arrb = (float*) nxt((size_t)B_ * H_ * S_ * 4);
    int*    idxb = (int*)   nxt((size_t)B_ * H_ * S_ * 4);
    bf16_t* h1b  = (bf16_t*)nxt((size_t)MS * 4 * E_ * 2);
    float*  hb   = (float*) nxt((size_t)MS * E_ * 4);

    // 1) bf16 conversions
    cvt_f32_bf16<<<dim3(MS * E_ / 1024), 256, 0, stream>>>(x, xb, MS * E_);
    cvt_f32_bf16<<<dim3(E_ * E_ / 1024), 256, 0, stream>>>(Wq, Wqb, E_ * E_);
    cvt_f32_bf16<<<dim3(E_ * E_ / 1024), 256, 0, stream>>>(Wk, Wkb, E_ * E_);
    cvt_f32_bf16<<<dim3(E_ * E_ / 1024), 256, 0, stream>>>(Wv, Wvb, E_ * E_);
    cvt_f32_bf16<<<dim3(4 * E_ * E_ / 1024), 256, 0, stream>>>(W1, W1b, 4 * E_ * E_);
    cvt_f32_bf16<<<dim3(4 * E_ * E_ / 1024), 256, 0, stream>>>(W2, W2b, 4 * E_ * E_);

    // 2) QKV projections (WMMA, async double-buffered)
    gemm_bf16<false, true, false><<<dim3(E_ / 128, MS / 128), 256, 0, stream>>>(
        xb, Wqb, bq, Qb, nullptr, MS, E_, E_);
    gemm_bf16<false, true, false><<<dim3(E_ / 128, MS / 128), 256, 0, stream>>>(
        xb, Wkb, bk, Kb, nullptr, MS, E_, E_);
    gemm_bf16<false, true, false><<<dim3(E_ / 128, MS / 128), 256, 0, stream>>>(
        xb, Wvb, bv, Vb, nullptr, MS, E_, E_);

    // 3) flash attention + (1/l, argmax) per row (WMMA)
    flash_attn<<<dim3(S_ / 128, B_ * H_), 256, 0, stream>>>(Qb, Kb, Vb, Of, valsb, idxb);

    // 4) pot-block seq-linear + exact GELU
    pot_gemm<<<dim3(B_ * H_), 256, 0, stream>>>(valsb, Wp, bp, arrb);

    // 5) rank-1 argmax-overwrite correction, emit bf16 O for FFN
    pot_correct<<<dim3((B_ * H_ * S_ * 64) / 256), 256, 0, stream>>>(
        valsb, arrb, idxb, Vb, Of, Obf);

    // 6) FFN1 with exact GELU (WMMA), 7) FFN2 (WMMA, f32 out)
    gemm_bf16<true, true, false><<<dim3(4 * E_ / 128, MS / 128), 256, 0, stream>>>(
        Obf, W1b, b1, h1b, nullptr, MS, 4 * E_, E_);
    gemm_bf16<false, false, true><<<dim3(E_ / 128, MS / 128), 256, 0, stream>>>(
        h1b, W2b, b2, nullptr, hb, MS, E_, 4 * E_);

    // 8) LayerNorm -> output
    layernorm<<<dim3(MS), 256, 0, stream>>>(hb, gamma, beta, outp);
}
